// AttentionWithRelPos_24550033064661
// MI455X (gfx1250) — matrix operand order repaired and checked by reading, not compiled, not run on hardware
//
#include <hip/hip_runtime.h>

#define NTOK  577
#define NPAD  640
#define BDIM  384
#define HEADS 6
#define HD    64
#define BATCH 16
#define RPD2  2209     // 47*47
#define PATCH 576
#define NEGMAX (-3.402823466e+38f)

typedef __attribute__((ext_vector_type(8)))  float    v8f;
typedef __attribute__((ext_vector_type(16))) _Float16 v16h;
typedef __attribute__((ext_vector_type(4)))  int      v4i;

typedef __attribute__((address_space(1))) v4i* gptr_v4i;
typedef __attribute__((address_space(3))) v4i* lptr_v4i;

union Frag16 { v16h v; unsigned int u[8]; };

__device__ inline v8f zero8() {
  v8f z;
#pragma unroll
  for (int i = 0; i < 8; ++i) z[i] = 0.0f;
  return z;
}

__device__ inline unsigned long long pack4h(float a, float b, float c, float d) {
  union { _Float16 h[4]; unsigned long long u; } p;
  p.h[0] = (_Float16)a; p.h[1] = (_Float16)b;
  p.h[2] = (_Float16)c; p.h[3] = (_Float16)d;
  return p.u;
}

// ---- CDNA5 async global->LDS copy (16 bytes per lane) ----------------------
__device__ inline void async_copy16(const _Float16* gsrc, _Float16* ldst) {
#if __has_builtin(__builtin_amdgcn_global_load_async_to_lds_b128)
  __builtin_amdgcn_global_load_async_to_lds_b128(
      (gptr_v4i)gsrc, (lptr_v4i)ldst, 0, 0);
#else
  unsigned long long ga = (unsigned long long)(uintptr_t)gsrc;
  unsigned int la = (unsigned int)(uintptr_t)ldst;  // low 32 bits = LDS offset
  asm volatile("global_load_async_to_lds_b128 %0, %1, off"
               :: "v"(la), "v"(ga) : "memory");
#endif
}

__device__ inline void wait_async0() {
#if __has_builtin(__builtin_amdgcn_s_wait_asynccnt)
  __builtin_amdgcn_s_wait_asynccnt(0);
#else
  asm volatile("s_wait_asynccnt 0x0" ::: "memory");
#endif
}

// A fragment 16x32 f16 from row-major LDS tile: lane m holds row (row0+m),
// K pair base per ISA 7.12.2: kb = (v<4 ? 2v : 8+2v) + 8*hi
__device__ inline v16h load_a(const _Float16* lds, int row0, int ld, int col0) {
  const int lane = threadIdx.x & 31;
  const int m = lane & 15, hi = lane >> 4;
  Frag16 f;
  const _Float16* base = lds + (row0 + m) * ld + col0 + hi * 8;
#pragma unroll
  for (int v = 0; v < 8; ++v) {
    const int kb = (v < 4) ? (2 * v) : (8 + 2 * v);
    f.u[v] = *(const unsigned int*)(base + kb);
  }
  return f.v;
}

// B fragment 32x16 f16 where B[kk,n] = src[row0+n][col0+kk] (pairs contiguous).
// ISA layout: VGPR v / lane n holds B[kb, n], B[kb+1, n], kb = (v<4 ? 2v : 8+2v) + 8*hi.
__device__ inline v16h load_bt(const _Float16* lds, int row0, int ld, int col0) {
  const int lane = threadIdx.x & 31;
  const int n = lane & 15, hi = lane >> 4;
  Frag16 f;
  const _Float16* base = lds + (row0 + n) * ld + col0 + hi * 8;
#pragma unroll
  for (int v = 0; v < 8; ++v)
    f.u[v] = *(const unsigned int*)(base + ((v < 4) ? (2 * v) : (8 + 2 * v)));
  return f.v;
}

__device__ inline v8f wmma_f16(v16h a, v16h b, v8f c) {
  return __builtin_amdgcn_wmma_f32_16x16x32_f16(false, a, false, b, (short)0, c,
                                                false, false);
}

// ---------------- Kernel 0: expand rel-pos bias table (H,576,576) -----------
__global__ __launch_bounds__(256) void bias_expand(
    const float* __restrict__ rel_pos, const int* __restrict__ rel_idx,
    float* __restrict__ bias_tab) {
  const int total = HEADS * PATCH * PATCH;
  int idx = blockIdx.x * 256 + threadIdx.x;
  if (idx < total) {
    const int h = idx / (PATCH * PATCH);
    const int rem = idx - h * (PATCH * PATCH);
    bias_tab[idx] = rel_pos[h * RPD2 + rel_idx[rem]];
  }
}

// ---------------- Kernel 1: QKV GEMM (9232x384 @ 384x1152^T + b) ------------
__global__ __launch_bounds__(128) void qkv_gemm(
    const float* __restrict__ x, const float* __restrict__ w,
    const float* __restrict__ bvec, _Float16* __restrict__ qkv) {
  __shared__ __align__(16) _Float16 Xs[64 * 40];
  __shared__ __align__(16) _Float16 Ws[64 * 40];
  const int tid = threadIdx.x;
  const int wid = tid >> 5;
  const int m0 = blockIdx.x * 64;
  const int j0 = blockIdx.y * 64;
  const int M = BATCH * NTOK;

  v8f acc[4];
#pragma unroll
  for (int t = 0; t < 4; ++t) acc[t] = zero8();

  for (int k0 = 0; k0 < BDIM; k0 += 32) {
    __syncthreads();
#pragma unroll
    for (int i = 0; i < 4; ++i) {  // X tile 64x32 f32 -> f16
      const int idx = tid + i * 128;
      const int r = idx >> 3, c = (idx & 7) * 4;
      const int gr = m0 + r;
      float4 xv = make_float4(0.f, 0.f, 0.f, 0.f);
      if (gr < M) xv = *(const float4*)(x + (size_t)gr * BDIM + k0 + c);
      *(unsigned long long*)(Xs + r * 40 + c) = pack4h(xv.x, xv.y, xv.z, xv.w);
    }
#pragma unroll
    for (int i = 0; i < 4; ++i) {  // W tile 64x32
      const int idx = tid + i * 128;
      const int r = idx >> 3, c = (idx & 7) * 4;
      float4 wv = *(const float4*)(w + (size_t)(j0 + r) * BDIM + k0 + c);
      *(unsigned long long*)(Ws + r * 40 + c) = pack4h(wv.x, wv.y, wv.z, wv.w);
    }
    __syncthreads();
    v16h a = load_a(Xs, wid * 16, 40, 0);
#pragma unroll
    for (int t = 0; t < 4; ++t) {
      v16h b = load_bt(Ws, t * 16, 40, 0);
      acc[t] = wmma_f16(a, b, acc[t]);
    }
  }

  const int lane = tid & 31, n = lane & 15, hi = lane >> 4;
#pragma unroll
  for (int t = 0; t < 4; ++t) {
    const int j = j0 + t * 16 + n;
    const float bj = bvec[j];
    const int which = j / BDIM;
    const int rem = j - which * BDIM;
    const int hh = rem >> 6, dd = rem & 63;
#pragma unroll
    for (int r = 0; r < 8; ++r) {
      const int gm = m0 + wid * 16 + r + hi * 8;
      if (gm < M) {
        const int b = gm / NTOK;
        const int nn = gm - b * NTOK;
        const size_t off =
            (((size_t)(which * BATCH + b) * HEADS + hh) * NPAD + nn) * HD + dd;
        qkv[off] = (_Float16)(acc[t][r] + bj);
      }
    }
  }
}

// ---------------- Kernel 2: flash attention with rel-pos bias + mask --------
__global__ __launch_bounds__(128) void attn_kernel(
    const _Float16* __restrict__ qkv, const float* __restrict__ bias_tab,
    const int* __restrict__ mask, const int* __restrict__ patch_flag,
    float* __restrict__ attn_out) {
  __shared__ __align__(16) _Float16 Qs[64 * 72];
  __shared__ __align__(16) _Float16 Ks[32 * 72];
  __shared__ __align__(16) _Float16 Vt[64 * 40];   // transposed: Vt[d][kk]
  __shared__ __align__(16) _Float16 Ps[4][16 * 40];

  const int tid = threadIdx.x;
  const int wid = tid >> 5;
  const int lane = tid & 31;
  const int n16 = lane & 15, hi = lane >> 4;

  const int bh = blockIdx.x;  // 0..95
  const int qt = blockIdx.y;  // 0..9
  const int b = bh / HEADS, h = bh % HEADS;
  const size_t headoff = ((size_t)b * HEADS + h) * NPAD * HD;
  const _Float16* qp = qkv + headoff;
  const _Float16* kp = qkv + (size_t)BATCH * HEADS * NPAD * HD + headoff;
  const _Float16* vp = qkv + (size_t)2 * BATCH * HEADS * NPAD * HD + headoff;
  const int qbase = qt * 64;
  const int patch = patch_flag[0];

#pragma unroll
  for (int i = 0; i < 4; ++i) {  // stage Q tile 64x64 via async copies
    const int idx = tid + i * 128;         // 512 chunks of 16B
    const int r = idx >> 3, c = (idx & 7) * 8;
    async_copy16(qp + (size_t)(qbase + r) * HD + c, Qs + r * 72 + c);
  }
  wait_async0();
  __syncthreads();
  const v16h qa0 = load_a(Qs, wid * 16, 72, 0);
  const v16h qa1 = load_a(Qs, wid * 16, 72, 32);

  v8f o[4];
#pragma unroll
  for (int t = 0; t < 4; ++t) o[t] = zero8();
  float mrow[8], lrow[8];
#pragma unroll
  for (int r = 0; r < 8; ++r) { mrow[r] = NEGMAX; lrow[r] = 0.0f; }

  for (int k0 = 0; k0 < NTOK; k0 += 32) {
    __syncthreads();
#pragma unroll
    for (int i = 0; i < 2; ++i) {  // stage K tile 32x64 via async copies
      const int idx = tid + i * 128;       // 256 chunks of 16B
      const int r = idx >> 3, c = (idx & 7) * 8;
      async_copy16(kp + (size_t)(k0 + r) * HD + c, Ks + r * 72 + c);
    }
#pragma unroll
    for (int i = 0; i < 4; ++i) {  // stage V transposed: Vt[d][kk]
      const int idx = tid + i * 128;
      const int r = idx >> 4, c = (idx & 15) * 4;
      union { unsigned long long u; _Float16 h4[4]; } p;
      p.u = *(const unsigned long long*)(vp + (size_t)(k0 + r) * HD + c);
      Vt[(c + 0) * 40 + r] = p.h4[0];
      Vt[(c + 1) * 40 + r] = p.h4[1];
      Vt[(c + 2) * 40 + r] = p.h4[2];
      Vt[(c + 3) * 40 + r] = p.h4[3];
    }
    wait_async0();
    __syncthreads();

    v8f s0 = zero8(), s1 = zero8();
    {  // S = Q K^T over d=0..63; K^T B-frags are contiguous-pair loads from Ks
      v16h kb = load_bt(Ks, 0, 72, 0);
      s0 = wmma_f16(qa0, kb, s0);
      kb = load_bt(Ks, 0, 72, 32);
      s0 = wmma_f16(qa1, kb, s0);
      kb = load_bt(Ks, 16, 72, 0);
      s1 = wmma_f16(qa0, kb, s1);
      kb = load_bt(Ks, 16, 72, 32);
      s1 = wmma_f16(qa1, kb, s1);
    }

#pragma unroll
    for (int r = 0; r < 8; ++r) {  // scale + rel-pos bias + mask
      const int q = qbase + wid * 16 + r + hi * 8;
      const int kg0 = k0 + n16, kg1 = k0 + 16 + n16;
      float e0 = s0[r] * 0.125f;
      float e1 = s1[r] * 0.125f;
      const bool qin = (q < NTOK);
      if (patch && qin && q >= 1) {
        const float* bt = bias_tab + ((size_t)h * PATCH + (q - 1)) * PATCH - 1;
        if (kg0 >= 1 && kg0 < NTOK) e0 += bt[kg0];
        if (kg1 >= 1 && kg1 < NTOK) e1 += bt[kg1];
      }
      const int mv0 =
          (qin && kg0 < NTOK) ? mask[((size_t)b * NTOK + q) * NTOK + kg0] : 0;
      const int mv1 =
          (qin && kg1 < NTOK) ? mask[((size_t)b * NTOK + q) * NTOK + kg1] : 0;
      if (mv0 == 0) e0 = NEGMAX;
      if (mv1 == 0) e1 = NEGMAX;
      s0[r] = e0;
      s1[r] = e1;
    }

#pragma unroll
    for (int r = 0; r < 8; ++r) {  // online softmax (16-lane row reductions)
      float mx = fmaxf(s0[r], s1[r]);
      mx = fmaxf(mx, __shfl_xor(mx, 1));
      mx = fmaxf(mx, __shfl_xor(mx, 2));
      mx = fmaxf(mx, __shfl_xor(mx, 4));
      mx = fmaxf(mx, __shfl_xor(mx, 8));
      const float mnew = fmaxf(mrow[r], mx);
      const float alpha = __expf(mrow[r] - mnew);
      const float p0 = __expf(s0[r] - mnew);
      const float p1 = __expf(s1[r] - mnew);
      float rs = p0 + p1;
      rs += __shfl_xor(rs, 1);
      rs += __shfl_xor(rs, 2);
      rs += __shfl_xor(rs, 4);
      rs += __shfl_xor(rs, 8);
      lrow[r] = lrow[r] * alpha + rs;
      mrow[r] = mnew;
      s0[r] = p0;
      s1[r] = p1;
#pragma unroll
      for (int t = 0; t < 4; ++t) o[t][r] *= alpha;
    }

    // P (16x32) C-layout -> A-layout via per-wave LDS (LDS is in-order per wave)
    _Float16* P = &Ps[wid][0];
#pragma unroll
    for (int r = 0; r < 8; ++r) {
      P[(r + hi * 8) * 40 + n16] = (_Float16)s0[r];
      P[(r + hi * 8) * 40 + 16 + n16] = (_Float16)s1[r];
    }
    __builtin_amdgcn_wave_barrier();
    const v16h pa = load_a(P, 0, 40, 0);
#pragma unroll
    for (int t = 0; t < 4; ++t) {  // O += P V, d-slices of 16
      v16h vb = load_bt(Vt, t * 16, 40, 0);
      o[t] = wmma_f16(pa, vb, o[t]);
    }
  }

#pragma unroll
  for (int r = 0; r < 8; ++r) {  // normalize + store f32 (B,N,384)
    const int q = qbase + wid * 16 + r + hi * 8;
    if (q < NTOK) {
      const float inv = 1.0f / lrow[r];
#pragma unroll
      for (int t = 0; t < 4; ++t)
        attn_out[((size_t)b * NTOK + q) * BDIM + h * HD + t * 16 + n16] =
            o[t][r] * inv;
    }
  }
}

// ---------------- Kernel 3: output projection (9232x384 @ 384x384^T + b) ----
__global__ __launch_bounds__(128) void proj_gemm(
    const float* __restrict__ a, const float* __restrict__ w,
    const float* __restrict__ bvec, float* __restrict__ out) {
  __shared__ __align__(16) _Float16 Xs[64 * 40];
  __shared__ __align__(16) _Float16 Ws[64 * 40];
  const int tid = threadIdx.x;
  const int wid = tid >> 5;
  const int m0 = blockIdx.x * 64;
  const int j0 = blockIdx.y * 64;
  const int M = BATCH * NTOK;

  v8f acc[4];
#pragma unroll
  for (int t = 0; t < 4; ++t) acc[t] = zero8();

  for (int k0 = 0; k0 < BDIM; k0 += 32) {
    __syncthreads();
#pragma unroll
    for (int i = 0; i < 4; ++i) {
      const int idx = tid + i * 128;
      const int r = idx >> 3, c = (idx & 7) * 4;
      const int gr = m0 + r;
      float4 xv = make_float4(0.f, 0.f, 0.f, 0.f);
      if (gr < M) xv = *(const float4*)(a + (size_t)gr * BDIM + k0 + c);
      *(unsigned long long*)(Xs + r * 40 + c) = pack4h(xv.x, xv.y, xv.z, xv.w);
    }
#pragma unroll
    for (int i = 0; i < 4; ++i) {
      const int idx = tid + i * 128;
      const int r = idx >> 3, c = (idx & 7) * 4;
      float4 wv = *(const float4*)(w + (size_t)(j0 + r) * BDIM + k0 + c);
      *(unsigned long long*)(Ws + r * 40 + c) = pack4h(wv.x, wv.y, wv.z, wv.w);
    }
    __syncthreads();
    v16h af = load_a(Xs, wid * 16, 40, 0);
#pragma unroll
    for (int t = 0; t < 4; ++t) {
      v16h bf = load_bt(Ws, t * 16, 40, 0);
      acc[t] = wmma_f16(af, bf, acc[t]);
    }
  }

  const int lane = tid & 31, n = lane & 15, hi = lane >> 4;
#pragma unroll
  for (int t = 0; t < 4; ++t) {
    const int j = j0 + t * 16 + n;
    const float bj = bvec[j];
#pragma unroll
    for (int r = 0; r < 8; ++r) {
      const int gm = m0 + wid * 16 + r + hi * 8;
      if (gm < M) out[(size_t)gm * BDIM + j] = acc[t][r] + bj;
    }
  }
}

extern "C" void kernel_launch(void* const* d_in, const int* in_sizes, int n_in,
                              void* d_out, int out_size, void* d_ws,
                              size_t ws_size, hipStream_t stream) {
  const float* x      = (const float*)d_in[0];
  const float* qkv_w  = (const float*)d_in[1];
  const float* qkv_b  = (const float*)d_in[2];
  const float* proj_w = (const float*)d_in[3];
  const float* proj_b = (const float*)d_in[4];
  const float* relpos = (const float*)d_in[5];
  const int* relidx   = (const int*)d_in[6];
  const int* maskp    = (const int*)d_in[7];
  const int* patch    = (const int*)d_in[8];
  float* out = (float*)d_out;

  const size_t qkv_bytes  = (size_t)3 * BATCH * HEADS * NPAD * HD * 2;  // 23.6MB
  const size_t attn_bytes = (size_t)BATCH * NTOK * BDIM * 4;            // 14.2MB
  _Float16* qkv = (_Float16*)d_ws;
  float* attn = (float*)((char*)d_ws + qkv_bytes);
  float* bias_tab = (float*)((char*)d_ws + qkv_bytes + attn_bytes);

  // zero the N-padding rows of q/k/v so attention tiles need no bounds checks
  (void)hipMemsetAsync(d_ws, 0, qkv_bytes, stream);

  dim3 blk(128, 1, 1);

  const int bias_total = HEADS * PATCH * PATCH;
  bias_expand<<<dim3((bias_total + 255) / 256, 1, 1), dim3(256, 1, 1), 0,
                stream>>>(relpos, relidx, bias_tab);

  dim3 g1((BATCH * NTOK + 63) / 64, (3 * BDIM) / 64, 1);  // 145 x 18
  qkv_gemm<<<g1, blk, 0, stream>>>(x, qkv_w, qkv_b, qkv);

  dim3 g2(BATCH * HEADS, NPAD / 64, 1);  // 96 x 10
  attn_kernel<<<g2, blk, 0, stream>>>(qkv, bias_tab, maskp, patch, attn);

  dim3 g3((BATCH * NTOK + 63) / 64, BDIM / 64, 1);  // 145 x 6
  proj_gemm<<<g3, blk, 0, stream>>>(attn, proj_w, proj_b, out);
}
